// TLP_54004918780287
// MI455X (gfx1250) — matrix operand order repaired and checked by reading, not compiled, not run on hardware
//
#include <hip/hip_runtime.h>
#include <hip/hip_bf16.h>
#include <math.h>

typedef __attribute__((ext_vector_type(16))) _Float16 v16h;
typedef __attribute__((ext_vector_type(8)))  float    v8f;

#define B_      4
#define C_      150
#define H_      512
#define W_      512
#define HS      85
#define WS      85
#define SH      6
#define SW      6
#define K_      512
#define CP      160          // padded class count for 16x16 tiling
#define NSITE   (HS*WS)      // 7225
#define N1      (C_*NSITE)   // 1,083,750 per batch
#define NB      512          // CG blocks per batch (fixed for determinism)
#define GRADBLK 64
#define GRADCNT 28560        // 4*85*84 == 4*84*85
#define CGITERS 25
#define EPSF    1e-6f
#define KEIMG   5.0f
#define TAUI    2.0f         // 1/TAU_S

// ---------------- reduction helpers (deterministic, power-of-2 block) -------
__device__ __forceinline__ float blockReduceSum(float v, float* sm) {
  int t = threadIdx.x;
  sm[t] = v; __syncthreads();
  for (int off = blockDim.x >> 1; off > 0; off >>= 1) {
    if (t < off) sm[t] += sm[t + off];
    __syncthreads();
  }
  float r = sm[0]; __syncthreads();
  return r;
}
__device__ __forceinline__ float blockReduceMax(float v, float* sm) {
  int t = threadIdx.x;
  sm[t] = v; __syncthreads();
  for (int off = blockDim.x >> 1; off > 0; off >>= 1) {
    if (t < off) sm[t] = fmaxf(sm[t], sm[t + off]);
    __syncthreads();
  }
  float r = sm[0]; __syncthreads();
  return r;
}
__device__ __forceinline__ float sumPartials(const float* p, int n, float* sm) {
  float v = 0.f;
  for (int i = threadIdx.x; i < n; i += blockDim.x) v += p[i];
  return blockReduceSum(v, sm);
}

// ---------------- 1) normalize text rows -> f16 (padded to CP rows) ---------
__global__ void k_norm_text(const float* __restrict__ T, _Float16* __restrict__ Tn) {
  __shared__ float sm[256];
  int c = blockIdx.x;
  float ss = 0.f;
  if (c < C_)
    for (int k = threadIdx.x; k < K_; k += 256) { float v = T[c*K_ + k]; ss += v*v; }
  float tot = blockReduceSum(ss, sm);
  float inv = (c < C_) ? 1.0f / fmaxf(sqrtf(tot), 1e-12f) : 0.0f;
  for (int k = threadIdx.x; k < K_; k += 256)
    Tn[(size_t)c*K_ + k] = (_Float16)((c < C_) ? T[c*K_ + k] * inv : 0.0f);
}

// ---------------- 2) Gram = Tn * Tn^T via WMMA f32_16x16x32_f16 -------------
__global__ void k_gram(const _Float16* __restrict__ Tn, float* __restrict__ G) {
  int lane = threadIdx.x;          // block = 1 wave (32 threads), EXEC all ones
  int half = lane >> 4;            // 0 or 1
  int mrow = lane & 15;
  int rowA = blockIdx.y * 16;
  int colB = blockIdx.x * 16;
  v8f acc = {};
  for (int kb = 0; kb < K_; kb += 32) {
    // A fragment: 16x32 f16. lane<16: K {0-7,16-23}; lane>=16: K {8-15,24-31}
    const _Float16* Ar = Tn + (size_t)(rowA + mrow) * K_ + kb + half * 8;
    v16h a;
#pragma unroll
    for (int q = 0; q < 8; ++q) { a[q] = Ar[q]; a[q + 8] = Ar[16 + q]; }
    // B fragment: 32x16, B[k][n] = Tn[colB+n][k]. lane<16: K {0-7,16-23}; lane>=16: K {8-15,24-31}
    const _Float16* Br = Tn + (size_t)(colB + mrow) * K_ + kb + half * 8;
    v16h b;
#pragma unroll
    for (int q = 0; q < 8; ++q) { b[q] = Br[q]; b[q + 8] = Br[16 + q]; }
    acc = __builtin_amdgcn_wmma_f32_16x16x32_f16(false, a, false, b, (short)0, acc,
                                                 false, false);
  }
#pragma unroll
  for (int r = 0; r < 8; ++r)
    G[(size_t)(rowA + half * 8 + r) * CP + (colB + mrow)] = acc[r];
}

// ---------------- 3) sdiag from row softmax + diag boost --------------------
__global__ void k_sdiag(const float* __restrict__ G, float* __restrict__ sdiag) {
  __shared__ float sm[256];
  int c = blockIdx.x;
  int t = threadIdx.x;
  float v = (t < C_) ? G[(size_t)c*CP + t] * TAUI : -1e30f;
  float m = blockReduceMax(v, sm);
  float e = (t < C_) ? expf(v - m) : 0.f;
  float sumexp = blockReduceSum(e, sm);
  float p = e / sumexp;
  float rowsum = blockReduceSum(p, sm);
  if (t == c) sdiag[c] = (p + 10.0f) / (rowsum + 10.0f + EPSF);
}

// ---------------- 4) 6x6 avg-pool of logits --------------------------------
__global__ void k_ds_logits(const float* __restrict__ logits, float* __restrict__ X) {
  int i = blockIdx.x * 256 + threadIdx.x;
  if (i >= B_ * C_ * NSITE) return;
  int xs = i % WS; int ys = (i / WS) % HS; int bc = i / NSITE; // bc = b*C_+c
  const float* src = logits + (size_t)bc * H_ * W_ + (size_t)(ys*SH) * W_ + xs*SW;
  float acc = 0.f;
#pragma unroll
  for (int r = 0; r < SH; ++r)
#pragma unroll
    for (int q = 0; q < SW; ++q) acc += src[r * W_ + q];
  X[i] = acc * (1.0f / 36.0f);
}

// ---------------- 5) per-site softmax -> probs, Cw --------------------------
__global__ void k_probs(const float* __restrict__ X, const float* __restrict__ sdiag,
                        float* __restrict__ probs, float* __restrict__ Cw) {
  int i = blockIdx.x * 256 + threadIdx.x;
  if (i >= B_ * NSITE) return;
  int b = i / NSITE; int s = i % NSITE;
  const float* L = X + (size_t)b * N1 + s;
  float m = -1e30f;
  for (int c = 0; c < C_; ++c) m = fmaxf(m, L[(size_t)c * NSITE]);
  float sum = 0.f;
  for (int c = 0; c < C_; ++c) sum += expf(L[(size_t)c * NSITE] - m);
  float inv = 1.0f / sum;
  float agree = 0.f;
  float* P = probs + (size_t)b * N1 + s;
  for (int c = 0; c < C_; ++c) {
    float p = expf(L[(size_t)c * NSITE] - m) * inv;
    P[(size_t)c * NSITE] = p;
    agree += sdiag[c] * p * p;
  }
  agree = fminf(fmaxf(agree, 0.f), 1.f);
  float conf = fmaxf(inv, 0.05f);           // max prob == exp(0)/sum
  Cw[i] = conf * conf * (1.0f + agree);
}

// ---------------- 6) gray downsample ----------------------------------------
__global__ void k_gray(const float* __restrict__ image, float* __restrict__ grayS) {
  int i = blockIdx.x * 256 + threadIdx.x;
  if (i >= B_ * NSITE) return;
  int xs = i % WS; int ys = (i / WS) % HS; int b = i / NSITE;
  const float* img = image + (size_t)b * 3 * H_ * W_;
  size_t base = (size_t)(ys*SH) * W_ + xs*SW;
  float r = 0.f, g = 0.f, bl = 0.f;
#pragma unroll
  for (int rr = 0; rr < SH; ++rr)
#pragma unroll
    for (int q = 0; q < SW; ++q) {
      size_t o = base + rr * W_ + q;
      r  += img[o];
      g  += img[(size_t)H_*W_ + o];
      bl += img[(size_t)2*H_*W_ + o];
    }
  grayS[i] = 0.299f*(r*(1.f/36.f)) + 0.587f*(g*(1.f/36.f)) + 0.114f*(bl*(1.f/36.f));
}

// ---------------- 7) gradient sums (for gx/gy means) ------------------------
__global__ void k_gradsum(const float* __restrict__ grayS,
                          float* __restrict__ pGx, float* __restrict__ pGy) {
  __shared__ float sm[256];
  float sx = 0.f, sy = 0.f;
  for (int i = blockIdx.x * 256 + threadIdx.x; i < GRADCNT; i += GRADBLK * 256) {
    int xg = i % (WS-1); int ys = (i/(WS-1)) % HS; int b = i/((WS-1)*HS);
    const float* gr = grayS + (size_t)b * NSITE + ys * WS;
    sx += fabsf(gr[xg+1] - gr[xg]);
    int xs = i % WS; int yg = (i/WS) % (HS-1); int b2 = i/(WS*(HS-1));
    const float* g2 = grayS + (size_t)b2 * NSITE;
    sy += fabsf(g2[(yg+1)*WS + xs] - g2[yg*WS + xs]);
  }
  float rx = blockReduceSum(sx, sm);
  float ry = blockReduceSum(sy, sm);
  if (threadIdx.x == 0) { pGx[blockIdx.x] = rx; pGy[blockIdx.x] = ry; }
}

// ---------------- 8) edge weights w_h, w_v ----------------------------------
__global__ void k_weights(const float* __restrict__ grayS, const float* __restrict__ probs,
                          const float* __restrict__ sdiag,
                          const float* __restrict__ pGx, const float* __restrict__ pGy,
                          float* __restrict__ wh, float* __restrict__ wv) {
  __shared__ float sm[256];
  float gxm = sumPartials(pGx, GRADBLK, sm) * (1.0f / GRADCNT);
  float gym = sumPartials(pGy, GRADBLK, sm) * (1.0f / GRADCNT);
  int i = blockIdx.x * 256 + threadIdx.x;
  if (i >= B_ * NSITE) return;
  int xs = i % WS; int ys = (i / WS) % HS; int b = i / NSITE;
  int s = i - b * NSITE;
  bool hx = (xs > 0), hy = (ys > 0);
  float gh = 0.f, gv = 0.f;
  const float* P = probs + (size_t)b * N1 + s;
  for (int c = 0; c < C_; ++c) {
    float pc = P[(size_t)c * NSITE];
    float sd = sdiag[c];
    if (hx) gh += sd * pc * P[(size_t)c * NSITE - 1];
    if (hy) gv += sd * pc * P[(size_t)c * NSITE - WS];
  }
  float whv = 1.f, wvv = 1.f;
  if (hx) {
    float d = fabsf(grayS[i] - grayS[i-1]) / (gxm + EPSF);
    float g = fminf(fmaxf(gh, 0.f), 1.f);
    whv = fminf(fmaxf(expf(-KEIMG * d) * (1.f + g), 0.f), 1.f);
  }
  if (hy) {
    float d = fabsf(grayS[i] - grayS[i-WS]) / (gym + EPSF);
    float g = fminf(fmaxf(gv, 0.f), 1.f);
    wvv = fminf(fmaxf(expf(-KEIMG * d) * (1.f + g), 0.f), 1.f);
  }
  wh[i] = whv; wv[i] = wvv;
}

// ---------------- weighted Laplacian ----------------------------------------
__device__ __forceinline__ float lap5(const float* __restrict__ Xb,
                                      const float* __restrict__ whb,
                                      const float* __restrict__ wvb,
                                      int c, int ys, int xs) {
  const float* Xc = Xb + (size_t)c * NSITE;
  int s = ys * WS + xs;
  float x = Xc[s], out = 0.f;
  if (xs > 0)      out += whb[s]      * (x - Xc[s-1]);
  if (xs < WS-1)   out -= whb[s+1]    * (Xc[s+1] - x);
  if (ys > 0)      out += wvb[s]      * (x - Xc[s-WS]);
  if (ys < HS-1)   out -= wvb[s+WS]   * (Xc[s+WS] - x);
  return out;
}

// ---------------- 9) CG init: R = -lap(X0), P = R, partial(R.R) -------------
__global__ void k_cg_init(const float* __restrict__ X, const float* __restrict__ wh,
                          const float* __restrict__ wv,
                          float* __restrict__ R, float* __restrict__ P,
                          float* __restrict__ part) {
  __shared__ float sm[256];
  int b = blockIdx.y;
  const float* Xb  = X  + (size_t)b * N1;
  const float* whb = wh + (size_t)b * NSITE;
  const float* wvb = wv + (size_t)b * NSITE;
  float acc = 0.f;
  for (int i = blockIdx.x * 256 + threadIdx.x; i < N1; i += NB * 256) {
    int c = i / NSITE; int s = i % NSITE;
    float r = -lap5(Xb, whb, wvb, c, s / WS, s % WS);
    size_t idx = (size_t)b * N1 + i;
    R[idx] = r; P[idx] = r; acc += r * r;
  }
  float t = blockReduceSum(acc, sm);
  if (threadIdx.x == 0) part[b * NB + blockIdx.x] = t;
}

// ---------------- 10) AP = Cw*P + lap(P); partial(P.AP) ---------------------
__global__ void k_cg_ap(const float* __restrict__ P, const float* __restrict__ Cw,
                        const float* __restrict__ wh, const float* __restrict__ wv,
                        float* __restrict__ AP, float* __restrict__ partP) {
  __shared__ float sm[256];
  int b = blockIdx.y;
  const float* Pb  = P  + (size_t)b * N1;
  const float* whb = wh + (size_t)b * NSITE;
  const float* wvb = wv + (size_t)b * NSITE;
  const float* cwb = Cw + (size_t)b * NSITE;
  float acc = 0.f;
  for (int i = blockIdx.x * 256 + threadIdx.x; i < N1; i += NB * 256) {
    int c = i / NSITE; int s = i % NSITE;
    float pv = Pb[i];
    float ap = cwb[s] * pv + lap5(Pb, whb, wvb, c, s / WS, s % WS);
    AP[(size_t)b * N1 + i] = ap;
    acc += pv * ap;
  }
  float t = blockReduceSum(acc, sm);
  if (threadIdx.x == 0) partP[b * NB + blockIdx.x] = t;
}

// ---------------- 11) X += aP; R -= a*AP; partial(R.R) ----------------------
__global__ void k_cg_xr(const float* __restrict__ partP, const float* __restrict__ partOld,
                        const float* __restrict__ P, const float* __restrict__ AP,
                        float* __restrict__ X, float* __restrict__ R,
                        float* __restrict__ partNew) {
  __shared__ float sm[256];
  int b = blockIdx.y;
  float pap   = sumPartials(partP   + b * NB, NB, sm);
  float rsold = sumPartials(partOld + b * NB, NB, sm);
  float alpha = rsold / (pap + 1e-12f);
  size_t base = (size_t)b * N1;
  float acc = 0.f;
  for (int i = blockIdx.x * 256 + threadIdx.x; i < N1; i += NB * 256) {
    size_t idx = base + i;
    X[idx] += alpha * P[idx];
    float r = R[idx] - alpha * AP[idx];
    R[idx] = r; acc += r * r;
  }
  float t = blockReduceSum(acc, sm);
  if (threadIdx.x == 0) partNew[b * NB + blockIdx.x] = t;
}

// ---------------- 12) P = R + beta*P ----------------------------------------
__global__ void k_cg_p(const float* __restrict__ partNew, const float* __restrict__ partOld,
                       const float* __restrict__ R, float* __restrict__ P) {
  __shared__ float sm[256];
  int b = blockIdx.y;
  float rsnew = sumPartials(partNew + b * NB, NB, sm);
  float rsold = sumPartials(partOld + b * NB, NB, sm);
  float beta = rsnew / (rsold + 1e-12f);
  size_t base = (size_t)b * N1;
  for (int i = blockIdx.x * 256 + threadIdx.x; i < N1; i += NB * 256) {
    size_t idx = base + i;
    P[idx] = R[idx] + beta * P[idx];
  }
}

// ---------------- 13) bilinear upsample 85 -> 512 (half-pixel) --------------
__global__ void k_upsample(const float* __restrict__ X, float* __restrict__ out) {
  int i = blockIdx.x * 256 + threadIdx.x;
  if (i >= B_ * C_ * H_ * W_) return;
  int x = i % W_; int y = (i / W_) % H_; int bc = i / (H_ * W_);
  const float* Xc = X + (size_t)bc * NSITE;
  float fy = (y + 0.5f) * ((float)HS / H_) - 0.5f;
  float fx = (x + 0.5f) * ((float)WS / W_) - 0.5f;
  int y0 = (int)floorf(fy); float ty = fy - (float)y0;
  int x0 = (int)floorf(fx); float tx = fx - (float)x0;
  int y0c = min(max(y0, 0), HS-1), y1c = min(max(y0+1, 0), HS-1);
  int x0c = min(max(x0, 0), WS-1), x1c = min(max(x0+1, 0), WS-1);
  float v00 = Xc[y0c*WS + x0c], v01 = Xc[y0c*WS + x1c];
  float v10 = Xc[y1c*WS + x0c], v11 = Xc[y1c*WS + x1c];
  out[i] = (1.f-ty)*((1.f-tx)*v00 + tx*v01) + ty*((1.f-tx)*v10 + tx*v11);
}

// ---------------- host ------------------------------------------------------
extern "C" void kernel_launch(void* const* d_in, const int* in_sizes, int n_in,
                              void* d_out, int out_size, void* d_ws, size_t ws_size,
                              hipStream_t stream) {
  const float* image  = (const float*)d_in[0];
  const float* logits = (const float*)d_in[1];
  const float* text   = (const float*)d_in[2];
  float* out = (float*)d_out;

  char* base = (char*)d_ws;
  size_t off = 0;
  auto carve = [&](size_t bytes) -> void* {
    void* p = base + off;
    off += (bytes + 255) & ~(size_t)255;
    return p;
  };
  _Float16* Tn   = (_Float16*)carve((size_t)CP * K_ * sizeof(_Float16));
  float* G       = (float*)carve((size_t)CP * CP * 4);
  float* sdiag   = (float*)carve(256 * 4);
  float* grayS   = (float*)carve((size_t)B_ * NSITE * 4);
  float* Cw      = (float*)carve((size_t)B_ * NSITE * 4);
  float* wh      = (float*)carve((size_t)B_ * NSITE * 4);
  float* wv      = (float*)carve((size_t)B_ * NSITE * 4);
  float* pGx     = (float*)carve(GRADBLK * 4);
  float* pGy     = (float*)carve(GRADBLK * 4);
  float* partA   = (float*)carve((size_t)B_ * NB * 4);
  float* partB   = (float*)carve((size_t)B_ * NB * 4);
  float* partP   = (float*)carve((size_t)B_ * NB * 4);
  float* X       = (float*)carve((size_t)B_ * N1 * 4);
  float* Rv      = (float*)carve((size_t)B_ * N1 * 4);
  float* Pv      = (float*)carve((size_t)B_ * N1 * 4);
  float* APv     = (float*)carve((size_t)B_ * N1 * 4);
  float* probs   = APv;   // probs consumed before AP is first written

  k_norm_text<<<CP, 256, 0, stream>>>(text, Tn);
  k_gram<<<dim3(CP/16, CP/16), 32, 0, stream>>>(Tn, G);
  k_sdiag<<<C_, 256, 0, stream>>>(G, sdiag);

  int ndl = B_ * C_ * NSITE;
  k_ds_logits<<<(ndl + 255) / 256, 256, 0, stream>>>(logits, X);
  int nsite = B_ * NSITE;
  k_probs<<<(nsite + 255) / 256, 256, 0, stream>>>(X, sdiag, probs, Cw);
  k_gray<<<(nsite + 255) / 256, 256, 0, stream>>>(image, grayS);
  k_gradsum<<<GRADBLK, 256, 0, stream>>>(grayS, pGx, pGy);
  k_weights<<<(nsite + 255) / 256, 256, 0, stream>>>(grayS, probs, sdiag, pGx, pGy, wh, wv);

  dim3 cgGrid(NB, B_);
  k_cg_init<<<cgGrid, 256, 0, stream>>>(X, wh, wv, Rv, Pv, partA);
  float* partOld = partA;
  float* partNew = partB;
  for (int it = 0; it < CGITERS; ++it) {
    k_cg_ap<<<cgGrid, 256, 0, stream>>>(Pv, Cw, wh, wv, APv, partP);
    k_cg_xr<<<cgGrid, 256, 0, stream>>>(partP, partOld, Pv, APv, X, Rv, partNew);
    k_cg_p<<<cgGrid, 256, 0, stream>>>(partNew, partOld, Rv, Pv);
    float* t = partOld; partOld = partNew; partNew = t;
  }

  int nout = B_ * C_ * H_ * W_;
  k_upsample<<<(nout + 255) / 256, 256, 0, stream>>>(X, out);
}
